// SpikingMLP_72241349918861
// MI455X (gfx1250) — compile-verified, matrix operand order checked by reading, not run on hardware
//
#include <hip/hip_runtime.h>
#include <hip/hip_bf16.h>
#include <math.h>

// ---------------- problem constants ----------------
#define TT   4
#define BB   32
#define CC   384
#define HH   1536
#define NN   196
#define NPAD 224            // 7 tiles of 32 (each wave covers a 64h x 32n tile)
#define TB   (TT*BB)        // 128
#define TAU_INV (1.0f/1.5f)
#define VTH  1.0f
#define EPS  1e-5f

typedef __attribute__((ext_vector_type(16))) _Float16 v16h;
typedef __attribute__((ext_vector_type(8)))  _Float16 v8h;
typedef __attribute__((ext_vector_type(8)))  float    v8f;

// Build a 16-element f16 fragment from two contiguous 16-byte chunks
// (CDNA5 16-bit A/B VGPR packing: elems 0..7 = K base, 8..15 = K base+16).
__device__ __forceinline__ v16h load_frag(const _Float16* __restrict__ p) {
  v8h a = *(const v8h*)(p);
  v8h b = *(const v8h*)(p + 16);
  v16h r;
#pragma unroll
  for (int i = 0; i < 8; ++i) { r[i] = a[i]; r[i + 8] = b[i]; }
  return r;
}

__device__ __forceinline__ v8f wmma_f16(v16h a, v16h b, v8f c) {
  return __builtin_amdgcn_wmma_f32_16x16x32_f16(false, a, false, b, (short)0, c, false, false);
}

// ---------------- f32 -> f16 convert ----------------
__global__ void __launch_bounds__(256) k_cvt_f16(const float* __restrict__ src,
                                                 _Float16* __restrict__ dst, int n) {
  int i = blockIdx.x * 256 + threadIdx.x;
  if (i < n) dst[i] = (_Float16)src[i];
}

// ---------------- per-channel mean/rstd over (T,B,N) ----------------
// layout: x[(outer*nch + ch)*ninner + i], outer = T*B, ninner = N
__global__ void __launch_bounds__(256) k_stats_f32(const float* __restrict__ x,
                                                   float* __restrict__ stats,
                                                   int nch, int ninner, int nouter) {
  const int ch = blockIdx.x;
  const int total = nouter * ninner;
  float s = 0.f, s2 = 0.f;
  for (int j = threadIdx.x; j < total; j += 256) {
    int o = j / ninner, i = j - o * ninner;
    float v = x[((size_t)o * nch + ch) * ninner + i];
    s += v; s2 += v * v;
  }
  __shared__ float sh[512];
  sh[threadIdx.x] = s; sh[256 + threadIdx.x] = s2;
  __syncthreads();
  for (int st = 128; st > 0; st >>= 1) {
    if ((int)threadIdx.x < st) {
      sh[threadIdx.x] += sh[threadIdx.x + st];
      sh[256 + threadIdx.x] += sh[256 + threadIdx.x + st];
    }
    __syncthreads();
  }
  if (threadIdx.x == 0) {
    float inv = 1.f / (float)total;
    float mean = sh[0] * inv;
    float var  = sh[256] * inv - mean * mean;
    stats[ch]       = mean;
    stats[nch + ch] = rsqrtf(var + EPS);
  }
}

__global__ void __launch_bounds__(256) k_stats_f16(const _Float16* __restrict__ x,
                                                   float* __restrict__ stats,
                                                   int nch, int ninner, int nouter) {
  const int ch = blockIdx.x;
  const int total = nouter * ninner;
  float s = 0.f, s2 = 0.f;
  for (int j = threadIdx.x; j < total; j += 256) {
    int o = j / ninner, i = j - o * ninner;
    float v = (float)x[((size_t)o * nch + ch) * ninner + i];
    s += v; s2 += v * v;
  }
  __shared__ float sh[512];
  sh[threadIdx.x] = s; sh[256 + threadIdx.x] = s2;
  __syncthreads();
  for (int st = 128; st > 0; st >>= 1) {
    if ((int)threadIdx.x < st) {
      sh[threadIdx.x] += sh[threadIdx.x + st];
      sh[256 + threadIdx.x] += sh[256 + threadIdx.x + st];
    }
    __syncthreads();
  }
  if (threadIdx.x == 0) {
    float inv = 1.f / (float)total;
    float mean = sh[0] * inv;
    float var  = sh[256] * inv - mean * mean;
    stats[ch]       = mean;
    stats[nch + ch] = rsqrtf(var + EPS);
  }
}

// ---------------- BN1 + LIF1 + pack to s1t[(t*B+b)*NPAD + n][c] (f16) --------
__global__ void __launch_bounds__(256) k_pack1(const float* __restrict__ x,
                                               const float* __restrict__ stats,
                                               const float* __restrict__ gamma,
                                               const float* __restrict__ beta,
                                               _Float16* __restrict__ s1t) {
  size_t idx = (size_t)blockIdx.x * 256 + threadIdx.x;
  if (idx >= (size_t)BB * NPAD * CC) return;
  int c = (int)(idx % CC);
  size_t r = idx / CC;
  int n = (int)(r % NPAD);
  int b = (int)(r / NPAD);
  float rstd  = stats[CC + c];
  float scale = gamma[c] * rstd;
  float shift = beta[c] - stats[c] * scale;
  bool valid = (n < NN);
  float v = 0.f;
#pragma unroll
  for (int t = 0; t < TT; ++t) {
    float s = 0.f;
    if (valid) {
      float xv = x[(((size_t)t * BB + b) * CC + c) * NN + n];
      float xb = scale * xv + shift;
      v += (xb - v) * TAU_INV;
      s  = (v - VTH >= 0.f) ? 1.f : 0.f;
      v *= (1.f - s);
    }
    s1t[(((size_t)t * BB + b) * NPAD + n) * CC + c] = (_Float16)s;
  }
}

// ---------------- GEMM1: y1[tb][h][n] = sum_c w1[h][c]*s1[tb][n][c] + b1[h] --
// wave = one 64(h) x 32(n) tile: 8 accumulators, A fragments reused x2.
__global__ void __launch_bounds__(256) k_gemm1(const _Float16* __restrict__ w1h,   // HxC
                                               const _Float16* __restrict__ s1t,   // TB x NPAD x C
                                               const float* __restrict__ bias,
                                               _Float16* __restrict__ y1) {        // TB x H x N
  const int lane = threadIdx.x & 31;
  const int wave = blockIdx.x * 8 + (threadIdx.x >> 5);
  const int NT = NPAD / 32;        // 7
  const int HB = HH / 64;          // 24
  int tmp = wave;
  const int nT = tmp % NT; tmp /= NT;
  const int hB = tmp % HB; tmp /= HB;
  const int tb = tmp;              // 0..127 (grid sized exactly)
  const int n0 = nT * 32, h0 = hB * 64;
  const int l16 = lane & 15;
  const int khalf = (lane >> 4) << 3;          // 0 or 8

  const _Float16* brow0 = s1t + ((size_t)tb * NPAD + (n0 + l16)) * CC + khalf;
  const _Float16* brow1 = brow0 + (size_t)16 * CC;
  const _Float16* arow  = w1h + (size_t)(h0 + l16) * CC + khalf;

  v8f acc00 = {}, acc10 = {}, acc20 = {}, acc30 = {};
  v8f acc01 = {}, acc11 = {}, acc21 = {}, acc31 = {};
  for (int k0 = 0; k0 < CC; k0 += 32) {
    __builtin_prefetch(brow0 + k0 + 64, 0, 1);  // -> global_prefetch_b8
    __builtin_prefetch(brow1 + k0 + 64, 0, 1);
    v16h bf0 = load_frag(brow0 + k0);
    v16h bf1 = load_frag(brow1 + k0);
    v16h a0 = load_frag(arow + k0);
    v16h a1 = load_frag(arow + k0 + (size_t)16 * CC);
    v16h a2 = load_frag(arow + k0 + (size_t)32 * CC);
    v16h a3 = load_frag(arow + k0 + (size_t)48 * CC);
    acc00 = wmma_f16(a0, bf0, acc00);
    acc10 = wmma_f16(a1, bf0, acc10);
    acc20 = wmma_f16(a2, bf0, acc20);
    acc30 = wmma_f16(a3, bf0, acc30);
    acc01 = wmma_f16(a0, bf1, acc01);
    acc11 = wmma_f16(a1, bf1, acc11);
    acc21 = wmma_f16(a2, bf1, acc21);
    acc31 = wmma_f16(a3, bf1, acc31);
  }

  const int hbase = h0 + ((lane >> 4) << 3);   // D rows: M = r + 8*(lane>=16)
  const int nn0 = n0 + l16;
  const int nn1 = nn0 + 16;
  if (nn0 < NN) {
    const size_t obase = ((size_t)tb * HH + hbase) * NN + nn0;
#pragma unroll
    for (int r = 0; r < 8; ++r) {
      y1[obase + (size_t)(r)      * NN] = (_Float16)(acc00[r] + bias[hbase + r]);
      y1[obase + (size_t)(r + 16) * NN] = (_Float16)(acc10[r] + bias[hbase + r + 16]);
      y1[obase + (size_t)(r + 32) * NN] = (_Float16)(acc20[r] + bias[hbase + r + 32]);
      y1[obase + (size_t)(r + 48) * NN] = (_Float16)(acc30[r] + bias[hbase + r + 48]);
    }
  }
  if (nn1 < NN) {
    const size_t obase = ((size_t)tb * HH + hbase) * NN + nn1;
#pragma unroll
    for (int r = 0; r < 8; ++r) {
      y1[obase + (size_t)(r)      * NN] = (_Float16)(acc01[r] + bias[hbase + r]);
      y1[obase + (size_t)(r + 16) * NN] = (_Float16)(acc11[r] + bias[hbase + r + 16]);
      y1[obase + (size_t)(r + 32) * NN] = (_Float16)(acc21[r] + bias[hbase + r + 32]);
      y1[obase + (size_t)(r + 48) * NN] = (_Float16)(acc31[r] + bias[hbase + r + 48]);
    }
  }
}

// ---------------- BN2 + LIF2 + pack to s2t[(t*B+b)*NPAD + n][h] (f16) --------
__global__ void __launch_bounds__(256) k_pack2(const _Float16* __restrict__ y1,
                                               const float* __restrict__ stats,
                                               const float* __restrict__ gamma,
                                               const float* __restrict__ beta,
                                               _Float16* __restrict__ s2t) {
  size_t idx = (size_t)blockIdx.x * 256 + threadIdx.x;
  if (idx >= (size_t)BB * NPAD * HH) return;
  int h = (int)(idx % HH);
  size_t r = idx / HH;
  int n = (int)(r % NPAD);
  int b = (int)(r / NPAD);
  float rstd  = stats[HH + h];
  float scale = gamma[h] * rstd;
  float shift = beta[h] - stats[h] * scale;
  bool valid = (n < NN);
  float v = 0.f;
#pragma unroll
  for (int t = 0; t < TT; ++t) {
    float s = 0.f;
    if (valid) {
      float xv = (float)y1[(((size_t)t * BB + b) * HH + h) * NN + n];
      float xb = scale * xv + shift;
      v += (xb - v) * TAU_INV;
      s  = (v - VTH >= 0.f) ? 1.f : 0.f;
      v *= (1.f - s);
    }
    s2t[(((size_t)t * BB + b) * NPAD + n) * HH + h] = (_Float16)s;
  }
}

// ---------------- GEMM2: out[tb][c][n] = sum_h w2[c][h]*s2[tb][n][h] + b2[c] -
// wave = one 64(c) x 32(n) tile.
__global__ void __launch_bounds__(256) k_gemm2(const _Float16* __restrict__ w2h,   // CxH
                                               const _Float16* __restrict__ s2t,   // TB x NPAD x H
                                               const float* __restrict__ bias,
                                               float* __restrict__ out) {          // TB x C x N
  const int lane = threadIdx.x & 31;
  const int wave = blockIdx.x * 8 + (threadIdx.x >> 5);
  const int NT = NPAD / 32;        // 7
  const int CB = CC / 64;          // 6
  int tmp = wave;
  const int nT = tmp % NT; tmp /= NT;
  const int cB = tmp % CB; tmp /= CB;
  const int tb = tmp;
  const int n0 = nT * 32, c0 = cB * 64;
  const int l16 = lane & 15;
  const int khalf = (lane >> 4) << 3;

  const _Float16* brow0 = s2t + ((size_t)tb * NPAD + (n0 + l16)) * HH + khalf;
  const _Float16* brow1 = brow0 + (size_t)16 * HH;
  const _Float16* arow  = w2h + (size_t)(c0 + l16) * HH + khalf;

  v8f acc00 = {}, acc10 = {}, acc20 = {}, acc30 = {};
  v8f acc01 = {}, acc11 = {}, acc21 = {}, acc31 = {};
  for (int k0 = 0; k0 < HH; k0 += 32) {
    __builtin_prefetch(brow0 + k0 + 128, 0, 1);
    __builtin_prefetch(brow1 + k0 + 128, 0, 1);
    v16h bf0 = load_frag(brow0 + k0);
    v16h bf1 = load_frag(brow1 + k0);
    v16h a0 = load_frag(arow + k0);
    v16h a1 = load_frag(arow + k0 + (size_t)16 * HH);
    v16h a2 = load_frag(arow + k0 + (size_t)32 * HH);
    v16h a3 = load_frag(arow + k0 + (size_t)48 * HH);
    acc00 = wmma_f16(a0, bf0, acc00);
    acc10 = wmma_f16(a1, bf0, acc10);
    acc20 = wmma_f16(a2, bf0, acc20);
    acc30 = wmma_f16(a3, bf0, acc30);
    acc01 = wmma_f16(a0, bf1, acc01);
    acc11 = wmma_f16(a1, bf1, acc11);
    acc21 = wmma_f16(a2, bf1, acc21);
    acc31 = wmma_f16(a3, bf1, acc31);
  }

  const int cbase = c0 + ((lane >> 4) << 3);
  const int nn0 = n0 + l16;
  const int nn1 = nn0 + 16;
  if (nn0 < NN) {
    const size_t obase = ((size_t)tb * CC + cbase) * NN + nn0;
#pragma unroll
    for (int r = 0; r < 8; ++r) {
      out[obase + (size_t)(r)      * NN] = acc00[r] + bias[cbase + r];
      out[obase + (size_t)(r + 16) * NN] = acc10[r] + bias[cbase + r + 16];
      out[obase + (size_t)(r + 32) * NN] = acc20[r] + bias[cbase + r + 32];
      out[obase + (size_t)(r + 48) * NN] = acc30[r] + bias[cbase + r + 48];
    }
  }
  if (nn1 < NN) {
    const size_t obase = ((size_t)tb * CC + cbase) * NN + nn1;
#pragma unroll
    for (int r = 0; r < 8; ++r) {
      out[obase + (size_t)(r)      * NN] = acc01[r] + bias[cbase + r];
      out[obase + (size_t)(r + 16) * NN] = acc11[r] + bias[cbase + r + 16];
      out[obase + (size_t)(r + 32) * NN] = acc21[r] + bias[cbase + r + 32];
      out[obase + (size_t)(r + 48) * NN] = acc31[r] + bias[cbase + r + 48];
    }
  }
}

// ---------------- host launch ----------------
extern "C" void kernel_launch(void* const* d_in, const int* in_sizes, int n_in,
                              void* d_out, int out_size, void* d_ws, size_t ws_size,
                              hipStream_t stream) {
  const float* x         = (const float*)d_in[0];
  const float* fc1_w     = (const float*)d_in[1];
  const float* fc1_b     = (const float*)d_in[2];
  const float* fc1_gamma = (const float*)d_in[3];
  const float* fc1_beta  = (const float*)d_in[4];
  const float* fc2_w     = (const float*)d_in[5];
  const float* fc2_b     = (const float*)d_in[6];
  const float* fc2_gamma = (const float*)d_in[7];
  const float* fc2_beta  = (const float*)d_in[8];
  float* out = (float*)d_out;

  // workspace layout (bytes, 256-aligned)
  char* ws = (char*)d_ws;
  size_t off = 0;
  float*     stats1 = (float*)(ws + off);     off += ((size_t)2 * CC * 4 + 255) & ~255ull;
  float*     stats2 = (float*)(ws + off);     off += ((size_t)2 * HH * 4 + 255) & ~255ull;
  _Float16*  w1h    = (_Float16*)(ws + off);  off += ((size_t)HH * CC * 2 + 255) & ~255ull;
  _Float16*  w2h    = (_Float16*)(ws + off);  off += ((size_t)CC * HH * 2 + 255) & ~255ull;
  _Float16*  s1t    = (_Float16*)(ws + off);  off += ((size_t)TB * NPAD * CC * 2 + 255) & ~255ull;
  _Float16*  y1     = (_Float16*)(ws + off);  off += ((size_t)TB * HH * NN * 2 + 255) & ~255ull;
  _Float16*  s2t    = (_Float16*)(ws + off);  off += ((size_t)TB * NPAD * HH * 2 + 255) & ~255ull;
  (void)ws_size; (void)n_in; (void)in_sizes; (void)out_size;

  // 1) weights -> f16
  {
    int n = HH * CC;
    k_cvt_f16<<<(n + 255) / 256, 256, 0, stream>>>(fc1_w, w1h, n);
    k_cvt_f16<<<(n + 255) / 256, 256, 0, stream>>>(fc2_w, w2h, n);
  }
  // 2) BN1 stats (per C channel over T*B*N)
  k_stats_f32<<<CC, 256, 0, stream>>>(x, stats1, CC, NN, TB);
  // 3) BN1 + LIF1 + pack (transposed, c-contiguous)
  {
    size_t total = (size_t)BB * NPAD * CC;
    k_pack1<<<(unsigned)((total + 255) / 256), 256, 0, stream>>>(x, stats1, fc2_gamma, fc2_beta, s1t);
  }
  // 4) GEMM1 (WMMA): waves = TB * (H/64) * (NPAD/32) = 128*24*7 = 21504, 8 waves/block
  k_gemm1<<<21504 / 8, 256, 0, stream>>>(w1h, s1t, fc1_b, y1);
  // 5) BN2 stats (per H channel over T*B*N)
  k_stats_f16<<<HH, 256, 0, stream>>>(y1, stats2, HH, NN, TB);
  // 6) BN2 + LIF2 + pack
  {
    size_t total = (size_t)BB * NPAD * HH;
    k_pack2<<<(unsigned)((total + 255) / 256), 256, 0, stream>>>(y1, stats2, fc1_gamma, fc1_beta, s2t);
  }
  // 7) GEMM2 (WMMA): waves = TB * (C/64) * (NPAD/32) = 128*6*7 = 5376
  k_gemm2<<<5376 / 8, 256, 0, stream>>>(w2h, s2t, fc2_b, out);
}
